// AudioTransformer_57389353009235
// MI455X (gfx1250) — matrix operand order repaired and checked by reading, not compiled, not run on hardware
//
#include <hip/hip_runtime.h>
#include <math.h>

// ---------------- model constants ----------------
constexpr int B   = 8;
constexpr int S   = 256;
constexpr int IN  = 128;
constexpr int D   = 512;
constexpr int H   = 16;
constexpr int L   = 8;
constexpr int FF  = 2048;
constexpr int LAT = 256;
constexpr int DH  = D / H;      // 32
constexpr int BS  = B * S;      // 2048 token rows

typedef __attribute__((ext_vector_type(16))) _Float16 v16h;
typedef __attribute__((ext_vector_type(8)))  float    v8f;
typedef __attribute__((ext_vector_type(4)))  _Float16 h4;

__device__ __forceinline__ float gelu_f(float x) {
    return 0.5f * x * (1.0f + erff(x * 0.7071067811865476f));
}

// =====================================================================
// WMMA GEMM:  C[M,N] = act( A[M,K] @ W[N,K]^T + bias[N] )
//  A element (m,k) at A[m*as_m + k]          (K contiguous, 16B aligned)
//  W element (n,k) at W[n*ws_n + k*ws_k]     (WSK1: ws_k==1 vector path)
//  C element (m,n) at C[m*cs_m + n]
//  batched via blockIdx.z: off = (z/bdiv)*S1 + (z%bdiv)*S2 per operand.
// Requirements (guaranteed by the call sites):
//  - M % 64 == 0, K % 32 == 0 (no load guards needed)
//  - N may be < 64 (ctx GEMM, N=32): W-tile over-reads stay inside the
//    workspace allocation; epilogue guards stores with n < N.
// Block: 256 threads = 8 waves; tile 64x64; wave tile 16x32; K step 32.
// f16 operands (converted during LDS staging), f32 accumulate.
// =====================================================================
template <int ACT, bool WSK1>
__global__ __launch_bounds__(256) void gemm_wmma_kernel(
    const float* __restrict__ A, const float* __restrict__ W,
    const float* __restrict__ bias, float* __restrict__ C,
    int M, int N, int K,
    long as_m, long ws_n, long ws_k, long cs_m,
    int bdiv, long aS1, long aS2, long wS1, long wS2, long cS1, long cS2)
{
    const int z = blockIdx.z;
    A += (long)(z / bdiv) * aS1 + (long)(z % bdiv) * aS2;
    W += (long)(z / bdiv) * wS1 + (long)(z % bdiv) * wS2;
    C += (long)(z / bdiv) * cS1 + (long)(z % bdiv) * cS2;

    __shared__ alignas(16) _Float16 Asub[64][40];   // padded rows (80 B)
    __shared__ alignas(16) _Float16 Wsub[64][40];

    const int m0   = blockIdx.y * 64;
    const int n0   = blockIdx.x * 64;
    const int tid  = threadIdx.x;
    const int lane = tid & 31;
    const int wv   = tid >> 5;      // wave id 0..7
    const int wm   = wv & 3;        // row sub-tile (16 rows)
    const int wn   = wv >> 2;       // col sub-tile (32 cols)

    v8f acc0 = {};
    v8f acc1 = {};

    const int tr = tid >> 3;         // 0..31 (row within pass)
    const int tc = (tid & 7) * 4;    // 0..28 step 4

    for (int k0 = 0; k0 < K; k0 += 32) {
        __syncthreads();
        // cooperative staging: 64x32 fp32 -> f16 LDS tiles (vectorized)
        #pragma unroll
        for (int p = 0; p < 2; ++p) {
            const int r = p * 32 + tr;
            {   // A tile: contiguous along K
                const float4 av = *(const float4*)(A + (long)(m0 + r) * as_m + (k0 + tc));
                h4 hv = { (_Float16)av.x, (_Float16)av.y, (_Float16)av.z, (_Float16)av.w };
                *(h4*)(&Asub[r][tc]) = hv;
            }
            if (WSK1) {  // W tile: contiguous along K
                const float4 wv4 = *(const float4*)(W + (long)(n0 + r) * ws_n + (k0 + tc));
                h4 hv = { (_Float16)wv4.x, (_Float16)wv4.y, (_Float16)wv4.z, (_Float16)wv4.w };
                *(h4*)(&Wsub[r][tc]) = hv;
            } else {     // strided gather (ctx GEMM: W = V, ws_k = D)
                const long wb = (long)(n0 + r) * ws_n + (long)(k0 + tc) * ws_k;
                const float w0v = W[wb];
                const float w1v = W[wb + ws_k];
                const float w2v = W[wb + 2 * ws_k];
                const float w3v = W[wb + 3 * ws_k];
                h4 hv = { (_Float16)w0v, (_Float16)w1v, (_Float16)w2v, (_Float16)w3v };
                *(h4*)(&Wsub[r][tc]) = hv;
            }
        }
        __syncthreads();

        // build fragments per ISA VGPR layout (wave32)
        const int hs  = lane >> 4;        // half-select
        const int lm  = lane & 15;
        const int akb = hs * 8;           // A: K base (0 or 8), +16 for upper
        const int bkb = hs * 8;           // B: K base (0 or 8), +16 for upper

        v16h af; v16h bf0; v16h bf1;
        #pragma unroll
        for (int i = 0; i < 8; ++i) {
            af[i]     = Asub[wm * 16 + lm][akb + i];
            af[8 + i] = Asub[wm * 16 + lm][16 + akb + i];
        }
        const int bn0 = wn * 32 + lm;
        #pragma unroll
        for (int i = 0; i < 16; ++i) {
            bf0[i] = Wsub[bn0][bkb + i + (i >= 8 ? 8 : 0)];
            bf1[i] = Wsub[bn0 + 16][bkb + i + (i >= 8 ? 8 : 0)];
        }
        acc0 = __builtin_amdgcn_wmma_f32_16x16x32_f16(false, af, false, bf0,
                                                      (short)0, acc0, false, false);
        acc1 = __builtin_amdgcn_wmma_f32_16x16x32_f16(false, af, false, bf1,
                                                      (short)0, acc1, false, false);
    }

    // epilogue: C/D layout -> VGPR r: m = r + 8*(lane>=16), n = lane&15
    const int lm      = lane & 15;
    const int rowbase = m0 + wm * 16 + ((lane >> 4) * 8);
    #pragma unroll
    for (int t = 0; t < 2; ++t) {
        v8f accv = t ? acc1 : acc0;
        const int n = n0 + wn * 32 + t * 16 + lm;
        if (n < N) {
            const float bvv = bias ? bias[n] : 0.0f;
            #pragma unroll
            for (int r = 0; r < 8; ++r) {
                const int m = rowbase + r;
                float val = accv[r] + bvv;
                if (ACT == 1) val = gelu_f(val);
                C[(long)m * cs_m + n] = val;
            }
        }
    }
}

// ---------------- residual + LayerNorm (r may be nullptr; in-place ok) ----
__global__ __launch_bounds__(256) void ln_res_kernel(
    const float* __restrict__ x, const float* __restrict__ r,
    const float* __restrict__ g, const float* __restrict__ b,
    float* __restrict__ y, int C)
{
    __shared__ float reds[8], reds2[8];
    const long row = blockIdx.x;
    const float* xr = x + row * C;
    const float* rr = r ? r + row * C : nullptr;
    float s = 0.f, s2 = 0.f;
    for (int c = threadIdx.x; c < C; c += 256) {
        float vv = xr[c] + (rr ? rr[c] : 0.0f);
        s += vv; s2 += vv * vv;
    }
    for (int o = 16; o > 0; o >>= 1) { s += __shfl_down(s, o, 32); s2 += __shfl_down(s2, o, 32); }
    if ((threadIdx.x & 31) == 0) { reds[threadIdx.x >> 5] = s; reds2[threadIdx.x >> 5] = s2; }
    __syncthreads();
    if (threadIdx.x == 0) {
        float ts = 0.f, ts2 = 0.f;
        for (int i = 0; i < 8; ++i) { ts += reds[i]; ts2 += reds2[i]; }
        reds[0] = ts; reds2[0] = ts2;
    }
    __syncthreads();
    const float mean = reds[0] / C;
    const float var  = reds2[0] / C - mean * mean;
    const float rstd = rsqrtf(var + 1e-5f);
    for (int c = threadIdx.x; c < C; c += 256) {
        float vv = xr[c] + (rr ? rr[c] : 0.0f);
        y[row * C + c] = (vv - mean) * rstd * g[c] + b[c];
    }
}

// ---------------- softmax with scale + ALiBi (row length S = 256) ---------
__global__ __launch_bounds__(256) void softmax_alibi_kernel(
    float* __restrict__ sc, int use_alibi)
{
    __shared__ float red[8];
    const int q = blockIdx.x;
    const int z = blockIdx.y;          // b*H + h
    const int h = z & (H - 1);
    float* row = sc + ((long)z * S + q) * S;
    const int k = threadIdx.x;
    float v = row[k] * 0.17677669529663687f;        // DH^-0.5
    if (use_alibi)
        v -= exp2f(-0.5f * (float)(h + 1)) * fabsf((float)(q - k));
    float m = v;
    for (int o = 16; o > 0; o >>= 1) m = fmaxf(m, __shfl_down(m, o, 32));
    if ((k & 31) == 0) red[k >> 5] = m;
    __syncthreads();
    if (k == 0) { float t = red[0]; for (int i = 1; i < 8; ++i) t = fmaxf(t, red[i]); red[0] = t; }
    __syncthreads();
    const float mx = red[0];
    __syncthreads();
    const float e = expf(v - mx);
    float s = e;
    for (int o = 16; o > 0; o >>= 1) s += __shfl_down(s, o, 32);
    if ((k & 31) == 0) red[k >> 5] = s;
    __syncthreads();
    if (k == 0) { float t = 0.f; for (int i = 0; i < 8; ++i) t += red[i]; red[0] = t; }
    __syncthreads();
    row[k] = e / red[0];
}

// ---------------- RoPE on q and k, layout (b*S+s, h*DH+dh) ---------------
__global__ void rope_kernel(float* __restrict__ q, float* __restrict__ k, int total)
{
    const int idx = blockIdx.x * blockDim.x + threadIdx.x;
    if (idx >= total) return;
    const int half = DH / 2;                 // 16
    int j = idx % half; int t = idx / half;
    int h = t % H; t /= H;
    int s = t % S; int b = t / S;
    const float inv = powf(10000.0f, -(float)j / (float)half);
    const float ang = (float)s * inv;
    const float c = cosf(ang), sn = sinf(ang);
    const long base = ((long)(b * S + s)) * D + h * DH + j;
    float q1 = q[base], q2 = q[base + half];
    q[base]        = q1 * c - q2 * sn;
    q[base + half] = q1 * sn + q2 * c;
    float k1 = k[base], k2 = k[base + half];
    k[base]        = k1 * c - k2 * sn;
    k[base + half] = k1 * sn + k2 * c;
}

// ---------------- to_lat: lat[b,n] = gelu(h[b,:] . W[n,:] + bias[n]) -----
__global__ __launch_bounds__(256) void to_lat_kernel(
    const float* __restrict__ hsrc, const float* __restrict__ w,
    const float* __restrict__ bias, float* __restrict__ lat)
{
    const int Kb = S * D;                        // 131072
    const int n = blockIdx.x;
    const float* wr = w + (long)n * Kb;
    float acc[B] = {};
    for (int k = threadIdx.x * 4; k < Kb; k += 256 * 4) {
        const float4 wv = *(const float4*)(wr + k);
        #pragma unroll
        for (int b = 0; b < B; ++b) {
            const float4 hv = *(const float4*)(hsrc + (long)b * Kb + k);
            acc[b] += wv.x * hv.x + wv.y * hv.y + wv.z * hv.z + wv.w * hv.w;
        }
    }
    __shared__ float red[8][B];
    #pragma unroll
    for (int b = 0; b < B; ++b) {
        float s = acc[b];
        for (int o = 16; o > 0; o >>= 1) s += __shfl_down(s, o, 32);
        if ((threadIdx.x & 31) == 0) red[threadIdx.x >> 5][b] = s;
    }
    __syncthreads();
    if (threadIdx.x < B) {
        const int b = threadIdx.x;
        float s = 0.f;
        for (int i = 0; i < 8; ++i) s += red[i][b];
        lat[b * LAT + n] = gelu_f(s + bias[n]);
    }
}

// ---------------- from_lat: mem[b,j] = gelu(lat[b,:] . W[j,:] + bias[j]) --
__global__ __launch_bounds__(256) void from_lat_kernel(
    const float* __restrict__ lat, const float* __restrict__ w,
    const float* __restrict__ bias, float* __restrict__ mem)
{
    const int Nn = S * D;                        // 131072
    const int j = blockIdx.x * 256 + threadIdx.x;
    if (j >= Nn) return;
    const float* wr = w + (long)j * LAT;
    float acc[B] = {};
    for (int k = 0; k < LAT; k += 4) {
        const float4 wv = *(const float4*)(wr + k);
        #pragma unroll
        for (int b = 0; b < B; ++b) {
            acc[b] += wv.x * lat[b * LAT + k]     + wv.y * lat[b * LAT + k + 1]
                    + wv.z * lat[b * LAT + k + 2] + wv.w * lat[b * LAT + k + 3];
        }
    }
    const float bj = bias[j];
    #pragma unroll
    for (int b = 0; b < B; ++b) mem[(long)b * Nn + j] = gelu_f(acc[b] + bj);
}

// ---------------- broadcast query tokens ---------------------------------
__global__ void bcast_query_kernel(const float* __restrict__ qt,
                                   float* __restrict__ dst, int total, int per)
{
    const int idx = blockIdx.x * blockDim.x + threadIdx.x;
    if (idx < total) dst[idx] = qt[idx % per];
}

// =====================================================================
// host-side helpers
// =====================================================================
static void gemm(hipStream_t st, const float* A, const float* W, const float* bias,
                 float* C, int M, int N, int K,
                 long as_m, long ws_n, long ws_k, long cs_m, int act,
                 int nz = 1, int bdiv = 1,
                 long aS1 = 0, long aS2 = 0, long wS1 = 0, long wS2 = 0,
                 long cS1 = 0, long cS2 = 0)
{
    dim3 grid((N + 63) / 64, (M + 63) / 64, nz);
    dim3 blk(256);
    if (ws_k == 1) {
        if (act)
            gemm_wmma_kernel<1, true><<<grid, blk, 0, st>>>(A, W, bias, C, M, N, K,
                as_m, ws_n, ws_k, cs_m, bdiv, aS1, aS2, wS1, wS2, cS1, cS2);
        else
            gemm_wmma_kernel<0, true><<<grid, blk, 0, st>>>(A, W, bias, C, M, N, K,
                as_m, ws_n, ws_k, cs_m, bdiv, aS1, aS2, wS1, wS2, cS1, cS2);
    } else {
        gemm_wmma_kernel<0, false><<<grid, blk, 0, st>>>(A, W, bias, C, M, N, K,
            as_m, ws_n, ws_k, cs_m, bdiv, aS1, aS2, wS1, wS2, cS1, cS2);
    }
}

static void mha(hipStream_t st, const float* q_in, const float* kv_in,
                const float* w4, const float* b4, bool use_rope, bool use_alibi,
                float* q, float* k, float* v, float* ctx, float* scores, float* out)
{
    // projections: (BS,D) = (BS,D) @ (D,D)^T + b
    gemm(st, q_in,  w4,                 b4,         q, BS, D, D, D, D, 1, D, 0);
    gemm(st, kv_in, w4 + (long)D * D,   b4 + D,     k, BS, D, D, D, D, 1, D, 0);
    gemm(st, kv_in, w4 + 2L * D * D,    b4 + 2 * D, v, BS, D, D, D, D, 1, D, 0);
    if (use_rope) {
        const int total = B * S * H * (DH / 2);
        rope_kernel<<<(total + 255) / 256, 256, 0, st>>>(q, k, total);
    }
    // scores[b,h] (S,S) = Q (S,DH) @ K^T : batched, head-strided
    gemm(st, q, k, nullptr, scores, S, S, DH,
         /*as_m*/ D, /*ws_n*/ D, /*ws_k*/ 1, /*cs_m*/ S, 0,
         B * H, H,
         (long)S * D, (long)DH,           // A offsets per b, per h
         (long)S * D, (long)DH,           // W (K buffer) offsets
         (long)H * S * S, (long)S * S);   // C offsets
    softmax_alibi_kernel<<<dim3(S, B * H), 256, 0, st>>>(scores, use_alibi ? 1 : 0);
    // ctx[b,h] (S,DH) = P (S,S) @ V (S,DH) ; W(n=d,k)=v[k*D+d] -> ws_n=1, ws_k=D
    gemm(st, scores, v, nullptr, ctx, S, DH, S,
         /*as_m*/ S, /*ws_n*/ 1, /*ws_k*/ D, /*cs_m*/ D, 0,
         B * H, H,
         (long)H * S * S, (long)S * S,
         (long)S * D, (long)DH,
         (long)S * D, (long)DH);
    // out projection
    gemm(st, ctx, w4 + 3L * D * D, b4 + 3 * D, out, BS, D, D, D, D, 1, D, 0);
}

extern "C" void kernel_launch(void* const* d_in, const int* in_sizes, int n_in,
                              void* d_out, int out_size, void* d_ws, size_t ws_size,
                              hipStream_t stream)
{
    (void)in_sizes; (void)n_in; (void)out_size; (void)ws_size;
    int i = 0;
    const float* x          = (const float*)d_in[i++];
    const float* proj_w     = (const float*)d_in[i++];
    const float* proj_b     = (const float*)d_in[i++];
    const float* query_tok  = (const float*)d_in[i++];
    const float* enc_attn_w = (const float*)d_in[i++];
    const float* enc_attn_b = (const float*)d_in[i++];
    const float* enc_ln_g   = (const float*)d_in[i++];
    const float* enc_ln_b   = (const float*)d_in[i++];
    const float* enc_ff1_w  = (const float*)d_in[i++];
    const float* enc_ff1_b  = (const float*)d_in[i++];
    const float* enc_ff2_w  = (const float*)d_in[i++];
    const float* enc_ff2_b  = (const float*)d_in[i++];
    const float* to_lat_w   = (const float*)d_in[i++];
    const float* to_lat_b   = (const float*)d_in[i++];
    const float* norm_g     = (const float*)d_in[i++];
    const float* norm_b     = (const float*)d_in[i++];
    const float* from_lat_w = (const float*)d_in[i++];
    const float* from_lat_b = (const float*)d_in[i++];
    const float* dec_self_w = (const float*)d_in[i++];
    const float* dec_self_b = (const float*)d_in[i++];
    const float* dec_crs_w  = (const float*)d_in[i++];
    const float* dec_crs_b  = (const float*)d_in[i++];
    const float* dec_ln_g   = (const float*)d_in[i++];
    const float* dec_ln_b   = (const float*)d_in[i++];
    const float* dec_ff1_w  = (const float*)d_in[i++];
    const float* dec_ff1_b  = (const float*)d_in[i++];
    const float* dec_ff2_w  = (const float*)d_in[i++];
    const float* dec_ff2_b  = (const float*)d_in[i++];
    const float* fc_w       = (const float*)d_in[i++];
    const float* fc_b       = (const float*)d_in[i++];

    // workspace carve (floats); total ~84 MB
    // NOTE: vb must be followed by another live buffer (ctxb) because the
    // N=32 ctx GEMM intentionally over-reads its W tile by < 64 floats.
    float* ws = (float*)d_ws;
    float* hA     = ws; ws += (long)BS * D;
    float* hB     = ws; ws += (long)BS * D;
    float* qb     = ws; ws += (long)BS * D;
    float* kb     = ws; ws += (long)BS * D;
    float* vb     = ws; ws += (long)BS * D;
    float* ctxb   = ws; ws += (long)BS * D;
    float* attnb  = ws; ws += (long)BS * D;
    float* memb   = ws; ws += (long)BS * D;
    float* ffmid  = ws; ws += (long)BS * FF;
    float* scores = ws; ws += (long)B * H * S * S;
    float* lat    = ws; ws += (long)B * LAT;

    // ---- input projection: h = gelu(x @ proj_w^T + proj_b)
    gemm(stream, x, proj_w, proj_b, hA, BS, D, IN, IN, IN, 1, D, /*gelu*/1);

    float* cur = hA;
    float* nxt = hB;

    // ---- encoder
    for (int l = 0; l < L; ++l) {
        const float* w4 = enc_attn_w + (long)l * 4 * D * D;
        const float* b4 = enc_attn_b + (long)l * 4 * D;
        mha(stream, cur, cur, w4, b4, true, true, qb, kb, vb, ctxb, scores, attnb);
        ln_res_kernel<<<BS, 256, 0, stream>>>(cur, attnb,
            enc_ln_g + (long)l * 2 * D, enc_ln_b + (long)l * 2 * D, nxt, D);
        { float* t = cur; cur = nxt; nxt = t; }
        gemm(stream, cur, enc_ff1_w + (long)l * FF * D, enc_ff1_b + (long)l * FF,
             ffmid, BS, FF, D, D, D, 1, FF, /*gelu*/1);
        gemm(stream, ffmid, enc_ff2_w + (long)l * D * FF, enc_ff2_b + (long)l * D,
             attnb, BS, D, FF, FF, FF, 1, D, 0);
        ln_res_kernel<<<BS, 256, 0, stream>>>(cur, attnb,
            enc_ln_g + (long)l * 2 * D + D, enc_ln_b + (long)l * 2 * D + D, nxt, D);
        { float* t = cur; cur = nxt; nxt = t; }
    }

    // ---- latent bottleneck (bandwidth-bound, fp32)
    to_lat_kernel<<<LAT, 256, 0, stream>>>(cur, to_lat_w, to_lat_b, lat);
    ln_res_kernel<<<B, 256, 0, stream>>>(lat, nullptr, norm_g, norm_b, lat, LAT);
    from_lat_kernel<<<(S * D) / 256, 256, 0, stream>>>(lat, from_lat_w, from_lat_b, memb);

    // ---- decoder queries
    {
        const int total = BS * D;
        bcast_query_kernel<<<(total + 255) / 256, 256, 0, stream>>>(query_tok, cur, total, S * D);
    }

    // ---- decoder
    for (int l = 0; l < L; ++l) {
        const float* ws4 = dec_self_w + (long)l * 4 * D * D;
        const float* bs4 = dec_self_b + (long)l * 4 * D;
        mha(stream, cur, cur, ws4, bs4, true, true, qb, kb, vb, ctxb, scores, attnb);
        ln_res_kernel<<<BS, 256, 0, stream>>>(cur, attnb,
            dec_ln_g + (long)l * 3 * D, dec_ln_b + (long)l * 3 * D, nxt, D);
        { float* t = cur; cur = nxt; nxt = t; }

        const float* wc4 = dec_crs_w + (long)l * 4 * D * D;
        const float* bc4 = dec_crs_b + (long)l * 4 * D;
        mha(stream, cur, memb, wc4, bc4, false, false, qb, kb, vb, ctxb, scores, attnb);
        ln_res_kernel<<<BS, 256, 0, stream>>>(cur, attnb,
            dec_ln_g + (long)l * 3 * D + D, dec_ln_b + (long)l * 3 * D + D, nxt, D);
        { float* t = cur; cur = nxt; nxt = t; }

        gemm(stream, cur, dec_ff1_w + (long)l * FF * D, dec_ff1_b + (long)l * FF,
             ffmid, BS, FF, D, D, D, 1, FF, /*gelu*/1);
        gemm(stream, ffmid, dec_ff2_w + (long)l * D * FF, dec_ff2_b + (long)l * D,
             attnb, BS, D, FF, FF, FF, 1, D, 0);
        ln_res_kernel<<<BS, 256, 0, stream>>>(cur, attnb,
            dec_ln_g + (long)l * 3 * D + 2 * D, dec_ln_b + (long)l * 3 * D + 2 * D, nxt, D);
        { float* t = cur; cur = nxt; nxt = t; }
    }

    // ---- output head: out = gelu(q @ fc_w^T + fc_b)  (fp32 output)
    gemm(stream, cur, fc_w, fc_b, (float*)d_out, BS, IN, D, D, D, 1, IN, /*gelu*/1);
}
